// FeatureDynamics_45707041964610
// MI455X (gfx1250) — compile-verified, matrix operand order checked by reading, not run on hardware
//
#include <hip/hip_runtime.h>

#define B_    8
#define N_    256
#define D_    3
#define F_    32
#define NRBF  50
#define H_    128
#define NK    255            // neighbors per node (N-1)
#define KPAD  64             // padded K for the rbf GEMMs
#define EPS_  1e-6f
#define VXSTRIDE (N_*D_ + N_*F_)   // 8960 floats per batch row of v

typedef _Float16 v16h __attribute__((ext_vector_type(16)));
typedef float    v8f  __attribute__((ext_vector_type(8)));

// K index inside a 32-wide K-step for half h (0..15) of a 16-bit WMMA fragment.
// Per CDNA5 ISA 7.12.2: VGPR v holds K pair base 16*(v>=4) + 8*group + 2*(v&3).
__device__ __forceinline__ int kmap(int h, int group) {
  const int v = h >> 1, lo = h & 1;
  return ((v & 4) << 2) | (group << 3) | ((v & 3) << 1) | lo;
}

// B-matrix fragment: columns of W1 (row-major [K][H_]) for this lane's column.
__device__ __forceinline__ v16h load_wfrag(const float* __restrict__ W, int ldw,
                                           int kofs, int col, int group) {
  v16h w;
#pragma unroll
  for (int h = 0; h < 16; ++h) {
    const int k = kofs + kmap(h, group);
    w[h] = (_Float16)W[k * ldw + col];
  }
  return w;
}

// Same, but rows pre-scaled by 2*gamma*mu_k (zero outside the NRBF rows):
// bakes d(rbf)/dd = -2g*d*rbf + 2g*mu*rbf into a weight-only transform.
__device__ __forceinline__ v16h load_wfrag_mu(const float* __restrict__ W,
                                              const float* __restrict__ musp,
                                              float g2, int kofs, int col, int group) {
  v16h w;
#pragma unroll
  for (int h = 0; h < 16; ++h) {
    const int k = kofs + kmap(h, group);
    w[h] = (k < NRBF) ? (_Float16)(g2 * musp[k] * W[k * H_ + col]) : (_Float16)0.f;
  }
  return w;
}

// A-matrix fragment from an LDS tile [16][KPAD] (row-major, f16).
__device__ __forceinline__ v16h load_afrag(const _Float16* __restrict__ A,
                                           int kofs, int m, int group) {
  v16h a;
#pragma unroll
  for (int h = 0; h < 16; ++h) {
    const int k = kofs + kmap(h, group);
    a[h] = A[m * KPAD + k];
  }
  return a;
}

__device__ __forceinline__ v8f wmma_f16(v16h a, v16h b, v8f c) {
  return __builtin_amdgcn_wmma_f32_16x16x32_f16(false, a, false, b,
                                                (short)0, c, false, false);
}

// Fast (1-ulp v_rcp_f32) sigmoid — avoids IEEE div chains in the WMMA epilogue.
__device__ __forceinline__ float fast_sigmoid(float x) {
  return __builtin_amdgcn_rcpf(1.f + __expf(-x));
}

// Pure-VALU xor-shuffle within independent 16-lane rows (v_permlane16_b32).
__device__ __forceinline__ float pl16(float x, int lo, int hi) {
  const int v = __builtin_amdgcn_permlane16(__float_as_int(x), __float_as_int(x),
                                            lo, hi, false, false);
  return __int_as_float(v);
}
// Sum across the 16 lanes of each half-wave (butterfly).
__device__ __forceinline__ float rowsum16(float x) {
  x += pl16(x, 0x67452301, 0xEFCDAB89);   // xor 1
  x += pl16(x, 0x54761032, 0xDCFE98BA);   // xor 2
  x += pl16(x, 0x32107654, 0xBA98FEDC);   // xor 4
  x += pl16(x, 0xFEDCBA98, 0x76543210);   // xor 8
  return x;
}

__global__ __launch_bounds__(256)
void pair_mlp_wmma(const float* __restrict__ state,
                   const float* __restrict__ mus,  const float* __restrict__ gammap,
                   const float* __restrict__ s1W1, const float* __restrict__ s1b1,
                   const float* __restrict__ s1W2, const float* __restrict__ s1b2,
                   const float* __restrict__ s2W1, const float* __restrict__ s2b1,
                   const float* __restrict__ s2W2, const float* __restrict__ s2b2,
                   const float* __restrict__ fW1,  const float* __restrict__ fb1,
                   const float* __restrict__ fW2,  const float* __restrict__ fb2,
                   float* __restrict__ out, float* __restrict__ vx_ws,
                   float* __restrict__ trace_ws)
{
  __shared__ float rxs[N_], rys[N_], rzs[N_], dvs[N_];
  __shared__ float cs1[H_], cs2[H_], cfs[H_], hfs[H_];
  __shared__ float s1v[N_], s2v[N_], udv[N_];
  __shared__ _Float16 Arbf[16 * KPAD];
  __shared__ float feats_sh[F_], sfeat[F_], xi[3], musl[NRBF];
  __shared__ float pstage[3][8][16];
  __shared__ float redw[8][8];

  const int t    = threadIdx.x;
  const int lane = t & 31;
  const int wave = t >> 5;
  const int grp  = (lane >> 4) & 1;
  const int b    = blockIdx.x >> 8;        // / N_
  const int i    = blockIdx.x & (N_ - 1);
  const float gamma = gammap[0];
  const float g2    = 2.f * gamma;

  const float* xb = state + (size_t)b * VXSTRIDE;    // x[b]: [N,3]
  const float* fbp = xb + N_ * D_;                   // feats[b]: [N,32]

  if (t < 3)    xi[t]       = xb[i * D_ + t];
  if (t < F_)   feats_sh[t] = fbp[i * F_ + t];
  if (t < NRBF) musl[t]     = mus[t];
  __syncthreads();

  // --- geometry: r[k] = x_i - x_j, d[k] = sqrt(|r|^2 + eps) -------------------
  {
    const int k = t;
    float rrx = 0.f, rry = 0.f, rrz = 0.f, dd = 1.f;
    if (k < NK) {
      const int j = (k < i) ? k : k + 1;            // OFFD[i][k]
      rrx = xi[0] - xb[j * D_ + 0];
      rry = xi[1] - xb[j * D_ + 1];
      rrz = xi[2] - xb[j * D_ + 2];
      dd  = sqrtf(rrx * rrx + rry * rry + rrz * rrz + EPS_);
    }
    rxs[k] = rrx; rys[k] = rry; rzs[k] = rrz; dvs[k] = dd;
  }

  // --- softmax(feats_i) over F=32, done by wave 0 -----------------------------
  if (wave == 0) {
    const float v = feats_sh[lane];
    float m = v;
#pragma unroll
    for (int o = 16; o > 0; o >>= 1) m = fmaxf(m, __shfl_xor(m, o, 32));
    const float e = __expf(v - m);
    float s = e;
#pragma unroll
    for (int o = 16; o > 0; o >>= 1) s += __shfl_xor(s, o, 32);
    sfeat[lane] = e * __builtin_amdgcn_rcpf(s);
  }
  __syncthreads();

  // --- k-invariant part of the hidden pre-activations (feature columns) ------
  if (t < H_) {
    float a1 = s1b1[t], a2 = s2b1[t], af = fb1[t];
    for (int f = 0; f < F_; ++f) {
      const float fe = feats_sh[f];
      a1 += fe * (s1W1[(NRBF + f) * H_ + t] + s1W1[(NRBF + F_ + f) * H_ + t]);
      a2 += fe * (s2W1[(NRBF + f) * H_ + t] + s2W1[(NRBF + F_ + f) * H_ + t]);
      af += sfeat[f] * fW1[(NRBF + f) * H_ + t];
    }
    cs1[t] = a1; cs2[t] = a2; cfs[t] = af;
  }
  __syncthreads();

  // --- resident weight fragments: wave owns hidden columns [wave*16, +16) ----
  const int col = (wave << 4) | (lane & 15);
  const v16h w1a = load_wfrag(s1W1, H_,  0, col, grp);
  const v16h w1b = load_wfrag(s1W1, H_, 32, col, grp);
  const v16h w2a = load_wfrag(s2W1, H_,  0, col, grp);
  const v16h w2b = load_wfrag(s2W1, H_, 32, col, grp);
  const v16h wfa = load_wfrag(fW1,  H_,  0, col, grp);
  const v16h wfb = load_wfrag(fW1,  H_, 32, col, grp);
  const v16h wma = load_wfrag_mu(s1W1, musl, g2,  0, col, grp);  // 2g*mu*W1 rows
  const v16h wmb = load_wfrag_mu(s1W1, musl, g2, 32, col, grp);
  const float s1w2 = s1W2[col];
  const float s2w2 = s2W2[col];
  const float c1 = cs1[col], c2 = cs2[col], cfl = cfs[col];

  float hfacc = 0.f;   // sum_k silu(zf) for this lane's hidden column

  for (int mt = 0; mt < 16; ++mt) {
    // build the 16x64 f16 rbf tile for rows k = mt*16..mt*16+15 (one b64 store)
    {
      const int row = t >> 4;
      const int c0  = (t & 15) << 2;
      const int k   = (mt << 4) | row;
      const float dd = dvs[k];
      const bool valid = (k < NK);
      union { _Float16 h[4]; uint2 u2; } pk;
#pragma unroll
      for (int cc = 0; cc < 4; ++cc) {
        const int c = c0 + cc;
        float e = 0.f;
        if (valid && c < NRBF) {
          const float dm = dd - musl[c];
          e = __expf(-gamma * dm * dm);
        }
        pk.h[cc] = (_Float16)e;
      }
      *(uint2*)&Arbf[row * KPAD + c0] = pk.u2;
    }
    __syncthreads();

    const int m = lane & 15;
    const v16h a0 = load_afrag(Arbf,  0, m, grp);
    const v16h a1 = load_afrag(Arbf, 32, m, grp);
    // this half-wave's 8 row distances (contiguous, broadcast LDS reads)
    union { float4 v4[2]; float f[8]; } du;
    du.v4[0] = *(const float4*)&dvs[(mt << 4) + (grp << 3)];
    du.v4[1] = *(const float4*)&dvs[(mt << 4) + (grp << 3) + 4];

    v8f z1 = {}; z1 = wmma_f16(a0, w1a, z1); z1 = wmma_f16(a1, w1b, z1);
    v8f z2 = {}; z2 = wmma_f16(a0, w2a, z2); z2 = wmma_f16(a1, w2b, z2);
    v8f zf = {}; zf = wmma_f16(a0, wfa, zf); zf = wmma_f16(a1, wfb, zf);
    v8f zm = {}; zm = wmma_f16(a0, wma, zm); zm = wmma_f16(a1, wmb, zm);

#pragma unroll
    for (int v = 0; v < 8; ++v) {
      const int krow = (mt << 4) + v + (grp << 3);     // C/D: row = v + 8*group
      const float mask = (krow < NK) ? 1.f : 0.f;
      const float zz1 = z1[v] + c1;
      const float sg  = fast_sigmoid(zz1);
      float p1 = zz1 * sg * s1w2 * mask;                          // silu(z)*W2
      // U = rbf@(2g*mu*W1) - 2g*d*(rbf@W1)  ->  gradient of s1 wrt d
      const float uu = zm[v] - g2 * du.f[v] * z1[v];
      float pu = sg * (1.f + zz1 * (1.f - sg)) * s1w2 * uu * mask; // silu'(z)*W2*U
      const float zz2 = z2[v] + c2;
      float p2 = zz2 * fast_sigmoid(zz2) * s2w2 * mask;
      const float zzf = zf[v] + cfl;
      hfacc += zzf * fast_sigmoid(zzf) * mask;
      // reduce over the 16 hidden columns held by this half-wave (pure VALU)
      p1 = rowsum16(p1);
      p2 = rowsum16(p2);
      pu = rowsum16(pu);
      if ((lane & 15) == 0) {
        pstage[0][wave][(grp << 3) + v] = p1;
        pstage[1][wave][(grp << 3) + v] = p2;
        pstage[2][wave][(grp << 3) + v] = pu;
      }
    }
    __syncthreads();

    if (t < 48) {                                       // combine 8 wave partials
      const int s = t >> 4, row = t & 15;
      float acc = 0.f;
#pragma unroll
      for (int w = 0; w < 8; ++w) acc += pstage[s][w][row];
      const int k = (mt << 4) | row;
      if (s == 0)      s1v[k] = acc;
      else if (s == 1) s2v[k] = acc;
      else             udv[k] = acc;
    }
  }

  // fold the two half-wave hidden sums; wave w exclusively owns cols w*16..+15
  {
    const float other = __shfl_xor(hfacc, 16, 32);
    if (grp == 0) hfs[col] = hfacc + other;
  }
  __syncthreads();

  // --- per-k scalars and workgroup reductions --------------------------------
  float tr = 0.f, ax = 0.f, ay = 0.f, az = 0.f, wx = 0.f, wy = 0.f, wz = 0.f;
  if (t < NK) {
    const float s1k = s1v[t] + s1b2[0];
    const float s2k = s2v[t] + s2b2[0];
    const float dd  = dvs[t];
    const float r2d = (dd * dd - EPS_) * __builtin_amdgcn_rcpf(dd); // |r|^2 / d
    tr = udv[t] * r2d + 3.f * s1k;                      // gd + D*s1
    ax = rxs[t] * s1k; ay = rys[t] * s1k; az = rzs[t] * s1k;
    wx = rxs[t] * s2k; wy = rys[t] * s2k; wz = rzs[t] * s2k;
  }
#pragma unroll
  for (int o = 16; o > 0; o >>= 1) {
    tr += __shfl_xor(tr, o, 32); ax += __shfl_xor(ax, o, 32);
    ay += __shfl_xor(ay, o, 32); az += __shfl_xor(az, o, 32);
    wx += __shfl_xor(wx, o, 32); wy += __shfl_xor(wy, o, 32);
    wz += __shfl_xor(wz, o, 32);
  }
  if (lane == 0) {
    float* rp = redw[wave];
    rp[0] = tr; rp[1] = ax; rp[2] = ay; rp[3] = az;
    rp[4] = wx; rp[5] = wy; rp[6] = wz;
  }
  __syncthreads();
  if (t == 0) {
    float T = 0, AX = 0, AY = 0, AZ = 0, WX = 0, WY = 0, WZ = 0;
    for (int w = 0; w < 8; ++w) {
      T += redw[w][0]; AX += redw[w][1]; AY += redw[w][2]; AZ += redw[w][3];
      WX += redw[w][4]; WY += redw[w][5]; WZ += redw[w][6];
    }
    const float inv = 1.f / (float)NK;
    const float a0 = AX * inv, a1c = AY * inv, a2c = AZ * inv;   // mean_k r*s1
    const float w0 = WX * inv, w1c = WY * inv, w2c = WZ * inv;   // mean_k r*s2
    const int base = (b * N_ + i) * 3;
    vx_ws[base + 0] = a0  + (w1c * a2c - w2c * a1c);   // a + w x a
    vx_ws[base + 1] = a1c + (w2c * a0  - w0  * a2c);
    vx_ws[base + 2] = a2c + (w0  * a1c - w1c * a0);
    trace_ws[b * N_ + i] = T;
  }
  // feature dynamics: (mean_k silu(zf)) @ fW2 + fb2
  if (t < F_) {
    const float inv = 1.f / (float)NK;
    float acc = fb2[t];
    for (int h = 0; h < H_; ++h) acc += hfs[h] * inv * fW2[h * F_ + t];
    out[(size_t)b * VXSTRIDE + N_ * D_ + i * F_ + t] = acc;
  }
}

__global__ __launch_bounds__(256)
void finalize_kernel(const float* __restrict__ vx_ws,
                     const float* __restrict__ trace_ws,
                     float* __restrict__ out)
{
  __shared__ float redw[8][4];
  __shared__ float tot[4];
  const int b = blockIdx.x, t = threadIdx.x;
  const int lane = t & 31, wave = t >> 5;
  const int base = (b * N_ + t) * 3;
  const float v0 = vx_ws[base], v1 = vx_ws[base + 1], v2 = vx_ws[base + 2];
  float s0 = v0, s1 = v1, s2 = v2, st = trace_ws[b * N_ + t];
#pragma unroll
  for (int o = 16; o > 0; o >>= 1) {
    s0 += __shfl_xor(s0, o, 32); s1 += __shfl_xor(s1, o, 32);
    s2 += __shfl_xor(s2, o, 32); st += __shfl_xor(st, o, 32);
  }
  if (lane == 0) {
    redw[wave][0] = s0; redw[wave][1] = s1; redw[wave][2] = s2; redw[wave][3] = st;
  }
  __syncthreads();
  if (t == 0) {
    float a = 0, bb = 0, c = 0, d = 0;
    for (int w = 0; w < 8; ++w) {
      a += redw[w][0]; bb += redw[w][1]; c += redw[w][2]; d += redw[w][3];
    }
    tot[0] = a / (float)N_; tot[1] = bb / (float)N_;
    tot[2] = c / (float)N_; tot[3] = d / (float)NK;     // trace /(N-1)
  }
  __syncthreads();
  float* ob = out + (size_t)b * VXSTRIDE;
  ob[t * 3 + 0] = v0 - tot[0];
  ob[t * 3 + 1] = v1 - tot[1];
  ob[t * 3 + 2] = v2 - tot[2];
  if (t == 0) out[(size_t)B_ * VXSTRIDE + b] = tot[3];
}

extern "C" void kernel_launch(void* const* d_in, const int* in_sizes, int n_in,
                              void* d_out, int out_size, void* d_ws, size_t ws_size,
                              hipStream_t stream)
{
  (void)in_sizes; (void)n_in; (void)out_size; (void)ws_size;
  // setup_inputs order: t, state, mus, gamma, s1_W1,s1_b1,s1_W2,s1_b2,
  //                     s2_W1,s2_b1,s2_W2,s2_b2, f_W1,f_b1,f_W2,f_b2
  const float* state = (const float*)d_in[1];
  const float* mus   = (const float*)d_in[2];
  const float* gamma = (const float*)d_in[3];
  const float* s1W1  = (const float*)d_in[4];
  const float* s1b1  = (const float*)d_in[5];
  const float* s1W2  = (const float*)d_in[6];
  const float* s1b2  = (const float*)d_in[7];
  const float* s2W1  = (const float*)d_in[8];
  const float* s2b1  = (const float*)d_in[9];
  const float* s2W2  = (const float*)d_in[10];
  const float* s2b2  = (const float*)d_in[11];
  const float* fW1   = (const float*)d_in[12];
  const float* fb1   = (const float*)d_in[13];
  const float* fW2   = (const float*)d_in[14];
  const float* fb2   = (const float*)d_in[15];

  float* out      = (float*)d_out;
  float* vx_ws    = (float*)d_ws;                 // B*N*3 floats
  float* trace_ws = vx_ws + B_ * N_ * D_;         // B*N floats

  pair_mlp_wmma<<<B_ * N_, 256, 0, stream>>>(state, mus, gamma,
      s1W1, s1b1, s1W2, s1b2, s2W1, s2b1, s2W2, s2b2,
      fW1, fb1, fW2, fb2, out, vx_ws, trace_ws);
  finalize_kernel<<<B_, 256, 0, stream>>>(vx_ws, trace_ws, out);
}
